// ImageSelfAttention_63393717289755
// MI455X (gfx1250) — matrix operand order repaired and checked by reading, not compiled, not run on hardware
//
#include <hip/hip_runtime.h>

// ---------------------------------------------------------------------------
// Image self-attention, fused flash-attention for gfx1250 (MI455X).
//   Kernel 1: QKV 1x1-conv projections in exact fp32, emitting bf16
//             Q^T [B][N][C] (pre-scaled 1/sqrt(C)), K^T [B][N][C], V [B][C][N].
//   Kernel 2: fused attention, 1 block = 4 wave32 = 64 query rows, looping
//             4096 keys in chunks of 64. K/V tiles are staged by the Tensor
//             Data Mover (tensor_load_to_lds) with double buffering: wave 0
//             issues the DMA for chunk i+1, waits s_wait_tensorcnt for chunk
//             i, and the TDM pad fields insert the 4-dword row padding that
//             makes B-operand ds reads bank-conflict-free. 16 bf16 WMMAs per
//             wave per chunk; the 512MB attention matrix never touches HBM.
// ---------------------------------------------------------------------------

typedef __attribute__((ext_vector_type(16))) __bf16 v16bf;
typedef __attribute__((ext_vector_type(8)))  __bf16 v8bf;
typedef __attribute__((ext_vector_type(8)))  float  v8f;
typedef __attribute__((ext_vector_type(4)))  unsigned int u32x4;
typedef __attribute__((ext_vector_type(8)))  int i32x8;
typedef __attribute__((ext_vector_type(4)))  int i32x4;

#define B_   8
#define C_   64
#define N_   4096
#define NPAD 72   // LDS row pitch in bf16: 144B = 36 dwords (32 data + 4 TDM pad)

// ---------------------------------------------------------------------------
// Kernel 1: QKV projection (exact fp32 math, bf16 outputs).
// grid = B * (N/64) = 512 blocks x 256 threads.
// ---------------------------------------------------------------------------
__global__ __launch_bounds__(256) void qkv_proj_kernel(
    const float* __restrict__ x,
    const float* __restrict__ Wq, const float* __restrict__ bq,
    const float* __restrict__ Wk, const float* __restrict__ bk,
    const float* __restrict__ Wv, const float* __restrict__ bv,
    __bf16* __restrict__ qT, __bf16* __restrict__ kT, __bf16* __restrict__ vCN)
{
    __shared__ float xs[C_][65];
    const int b  = blockIdx.x >> 6;
    const int n0 = (blockIdx.x & 63) << 6;
    const int t  = threadIdx.x;

    const float* xb = x + (size_t)b * C_ * N_;
#pragma unroll
    for (int i = 0; i < 16; ++i) {
        int c = (t >> 6) + i * 4;
        int j = t & 63;
        xs[c][j] = xb[(size_t)c * N_ + n0 + j];
    }
    __syncthreads();

    const int co = t & 63;
    float accq[16], acck[16], accv[16];
#pragma unroll
    for (int jj = 0; jj < 16; ++jj) { accq[jj] = 0.f; acck[jj] = 0.f; accv[jj] = 0.f; }

    const float* wqr = Wq + co * C_;
    const float* wkr = Wk + co * C_;
    const float* wvr = Wv + co * C_;
    for (int c = 0; c < C_; ++c) {
        const float wq_ = wqr[c], wk_ = wkr[c], wv_ = wvr[c];
#pragma unroll
        for (int jj = 0; jj < 16; ++jj) {
            float xv = xs[c][(t >> 6) + jj * 4];
            accq[jj] = fmaf(wq_, xv, accq[jj]);
            acck[jj] = fmaf(wk_, xv, acck[jj]);
            accv[jj] = fmaf(wv_, xv, accv[jj]);
        }
    }
    const float bqv = bq[co], bkv = bk[co], bvv = bv[co];
#pragma unroll
    for (int jj = 0; jj < 16; ++jj) {
        int n = n0 + (t >> 6) + jj * 4;
        size_t rc = ((size_t)b * N_ + n) * C_ + co;
        qT[rc] = (__bf16)((accq[jj] + bqv) * 0.125f);   // fold scale = 1/sqrt(64)
        kT[rc] = (__bf16)(acck[jj] + bkv);
        vCN[((size_t)b * C_ + co) * N_ + n] = (__bf16)(accv[jj] + bvv);
    }
}

// ---------------------------------------------------------------------------
// WMMA operand builders (wave32 layouts per CDNA5 ISA 7.12.2).
// ---------------------------------------------------------------------------
__device__ __forceinline__ v16bf make_a(const __bf16* p) {
    v8bf lo = *(const v8bf*)p;                   // K = off .. off+7
    v8bf hi = *(const v8bf*)(p + 16);            // K = off+16 .. off+23
    v16bf r;
#pragma unroll
    for (int i = 0; i < 8; ++i) { r[i] = lo[i]; r[8 + i] = hi[i]; }
    return r;
}
__device__ __forceinline__ v16bf make_b(const __bf16* p) {
    v8bf lo = *(const v8bf*)p;                   // K = off .. off+7
    v8bf hi = *(const v8bf*)(p + 16);            // K = off+16 .. off+23
    v16bf r;
#pragma unroll
    for (int i = 0; i < 8; ++i) { r[i] = lo[i]; r[8 + i] = hi[i]; }
    return r;
}

// ---------------------------------------------------------------------------
// Tensor Data Mover: 2-D tile DMA global->LDS with row padding.
// D# per CDNA5 ISA ch.8: group0 = {flags, lds_addr, global_addr, type=2},
// group1 = dims/strides + pad config (data_size=2B; pad 4 dwords after every
// 32 dwords -> 36-dword LDS row pitch = NPAD bf16). Groups 2/3 zero (2-D
// tile), trailing 8-dword group zero, cpol 0. This toolchain exposes the
// 6-argument tensor_load_to_lds builtin.
// ---------------------------------------------------------------------------
__device__ __forceinline__ void tdm_load_2d(
    const void* gaddr, unsigned lds_byte_addr,
    unsigned tensor_d0, unsigned tensor_d1, unsigned long long d0_stride,
    unsigned tile_d0, unsigned tile_d1)
{
    unsigned long long ga = (unsigned long long)(uintptr_t)gaddr;
    u32x4 g0;
    g0[0] = 1u;                                               // count=1 valid user D#
    g0[1] = lds_byte_addr;                                    // LDS dest (bytes)
    g0[2] = (unsigned)(ga & 0xFFFFFFFFu);                     // global_addr[31:0]
    g0[3] = (unsigned)((ga >> 32) & 0x01FFFFFFu) | 0x80000000u; // addr[56:32] | type=2
    i32x8 g1;
    g1[0] = (int)((1u << 16)      // data_size = 1 -> 2 bytes
                | (1u << 20)      // pad_enable
                | (4u << 22)      // pad_interval: 32 dwords between pads
                | (3u << 25));    // pad_amount:   4 dwords of pad
    g1[1] = (int)((tensor_d0 & 0xFFFFu) << 16);                       // dim0[15:0]
    g1[2] = (int)(((tensor_d0 >> 16) & 0xFFFFu)
                | ((tensor_d1 & 0xFFFFu) << 16));                     // dim0[31:16]|dim1[15:0]
    g1[3] = (int)(((tensor_d1 >> 16) & 0xFFFFu) | (tile_d0 << 16));   // dim1[31:16]|tile0
    g1[4] = (int)(tile_d1 & 0xFFFFu);                                 // tile1, tile2=0
    g1[5] = (int)(d0_stride & 0xFFFFFFFFu);                           // dim0_stride[31:0]
    g1[6] = (int)((d0_stride >> 32) & 0xFFFFu);                       // dim0_stride[47:32]
    g1[7] = 0;
    i32x4 z4 = { 0, 0, 0, 0 };
    i32x8 z8 = { 0, 0, 0, 0, 0, 0, 0, 0 };
    __builtin_amdgcn_tensor_load_to_lds(g0, g1, z4, z4, z8, 0);
}

// ---------------------------------------------------------------------------
// Kernel 2: fused flash attention with double-buffered TDM staging.
// grid = B * (N/64) = 512 blocks x 128 threads (4 wave32).
// ---------------------------------------------------------------------------
__global__ __launch_bounds__(128) void flash_attn_kernel(
    const __bf16* __restrict__ qT, const __bf16* __restrict__ kT,
    const __bf16* __restrict__ vCN, float* __restrict__ out)
{
    __shared__ __align__(16) __bf16 ks_s[2][C_ * NPAD];     // K^T tile: [m_local][c]
    __shared__ __align__(16) __bf16 vs_s[2][C_ * NPAD];     // V tile:   [c][m_local]
    __shared__ __align__(16) __bf16 ps_s[4 * 16 * NPAD];    // P per wave: [n_local][m_local]

    const int b     = blockIdx.x >> 6;
    const int n0    = (blockIdx.x & 63) << 6;
    const int t     = threadIdx.x;
    const int wave  = t >> 5;
    const int lane  = t & 31;
    const int lhalf = lane >> 4;
    const int lcol  = lane & 15;

    // Q A-operands for this wave's 16 query rows (scale pre-folded).
    const __bf16* qrow = qT + ((size_t)b * N_ + n0 + wave * 16 + lcol) * C_;
    const v16bf aq0 = make_a(qrow + 0 * 32 + lhalf * 8);
    const v16bf aq1 = make_a(qrow + 32 + lhalf * 8);

    v8f o[4] = {};
    float m8[8], l8[8];
#pragma unroll
    for (int r = 0; r < 8; ++r) { m8[r] = -3.0e38f; l8[r] = 0.f; }

    const __bf16* kTb = kT + (size_t)b * N_ * C_;
    const __bf16* vCb = vCN + (size_t)b * C_ * N_;

    // Prologue: wave 0 kicks off the DMA for the first K/V tile.
    if (wave == 0) {
        tdm_load_2d(kTb, (unsigned)(uintptr_t)&ks_s[0][0],
                    C_, N_, C_, /*tile*/ C_, 64);             // 64 key rows x 64 ch
        tdm_load_2d(vCb, (unsigned)(uintptr_t)&vs_s[0][0],
                    N_, C_, N_, /*tile*/ 64, C_);             // 64 ch rows x 64 keys
    }

    for (int m0 = 0; m0 < N_; m0 += 64) {
        const int buf = (m0 >> 6) & 1;
        __syncthreads();   // all waves done reading buf^1 from previous iteration
        if (wave == 0) {
            if (m0 + 64 < N_) {
                // Issue next tile into the other buffer, then wait until only
                // those 2 DMAs remain outstanding => current tile has landed.
                tdm_load_2d(kTb + (size_t)(m0 + 64) * C_,
                            (unsigned)(uintptr_t)&ks_s[buf ^ 1][0],
                            C_, N_, C_, C_, 64);
                tdm_load_2d(vCb + (m0 + 64),
                            (unsigned)(uintptr_t)&vs_s[buf ^ 1][0],
                            N_, C_, N_, 64, C_);
                __builtin_amdgcn_s_wait_tensorcnt(2);
            } else {
                __builtin_amdgcn_s_wait_tensorcnt(0);
            }
        }
        __syncthreads();   // tile `buf` visible to all waves
        const __bf16* ksb = &ks_s[buf][0];
        const __bf16* vsb = &vs_s[buf][0];

        // ---- S = (Q^T K): 4 key tiles of 16, K-dim = C = 64 -> 2 wmma each
        v8f s[4];
#pragma unroll
        for (int mt = 0; mt < 4; ++mt) {
            v8f acc = {};
            v16bf b0 = make_b(ksb + (mt * 16 + lcol) * NPAD + 0 * 32 + lhalf * 8);
            acc = __builtin_amdgcn_wmma_f32_16x16x32_bf16(false, aq0, false, b0,
                                                          (short)0, acc, false, false);
            v16bf b1 = make_b(ksb + (mt * 16 + lcol) * NPAD + 32 + lhalf * 8);
            acc = __builtin_amdgcn_wmma_f32_16x16x32_bf16(false, aq1, false, b1,
                                                          (short)0, acc, false, false);
            s[mt] = acc;
        }

        // ---- online softmax over this 64-key chunk
        float mx[8];
#pragma unroll
        for (int r = 0; r < 8; ++r)
            mx[r] = fmaxf(fmaxf(s[0][r], s[1][r]), fmaxf(s[2][r], s[3][r]));
#pragma unroll
        for (int d = 1; d < 16; d <<= 1)
#pragma unroll
            for (int r = 0; r < 8; ++r)
                mx[r] = fmaxf(mx[r], __shfl_xor(mx[r], d, 32));
        float corr[8];
#pragma unroll
        for (int r = 0; r < 8; ++r) {
            float mn = fmaxf(m8[r], mx[r]);
            corr[r]  = __expf(m8[r] - mn);
            m8[r]    = mn;
        }
        float rs[8];
#pragma unroll
        for (int r = 0; r < 8; ++r) rs[r] = 0.f;
#pragma unroll
        for (int mt = 0; mt < 4; ++mt)
#pragma unroll
            for (int r = 0; r < 8; ++r) {
                float p = __expf(s[mt][r] - m8[r]);
                s[mt][r] = p;
                rs[r] += p;
            }
#pragma unroll
        for (int d = 1; d < 16; d <<= 1)
#pragma unroll
            for (int r = 0; r < 8; ++r)
                rs[r] += __shfl_xor(rs[r], d, 32);
#pragma unroll
        for (int r = 0; r < 8; ++r) l8[r] = l8[r] * corr[r] + rs[r];
#pragma unroll
        for (int ct = 0; ct < 4; ++ct)
#pragma unroll
            for (int r = 0; r < 8; ++r) o[ct][r] *= corr[r];

        // ---- transpose P (D-layout -> A-layout) via per-wave LDS tile
        __bf16* pw = ps_s + wave * 16 * NPAD;
#pragma unroll
        for (int mt = 0; mt < 4; ++mt)
#pragma unroll
            for (int r = 0; r < 8; ++r)
                pw[(r + 8 * lhalf) * NPAD + mt * 16 + lcol] = (__bf16)s[mt][r];
        v16bf ap0 = make_a(pw + lcol * NPAD + 0 * 32 + lhalf * 8);
        v16bf ap1 = make_a(pw + lcol * NPAD + 32 + lhalf * 8);

        // ---- O^T += P * V^T : K-dim = 64 keys -> 2 wmma per 16-col c tile
#pragma unroll
        for (int ct = 0; ct < 4; ++ct) {
            v16bf b0 = make_b(vsb + (ct * 16 + lcol) * NPAD + 0 * 32 + lhalf * 8);
            o[ct] = __builtin_amdgcn_wmma_f32_16x16x32_bf16(false, ap0, false, b0,
                                                            (short)0, o[ct], false, false);
            v16bf b1 = make_b(vsb + (ct * 16 + lcol) * NPAD + 32 + lhalf * 8);
            o[ct] = __builtin_amdgcn_wmma_f32_16x16x32_bf16(false, ap1, false, b1,
                                                            (short)0, o[ct], false, false);
        }
    }

    // ---- normalize and store: out[b][c][n]; rows r are consecutive n
    float inv[8];
#pragma unroll
    for (int r = 0; r < 8; ++r) inv[r] = 1.0f / l8[r];
#pragma unroll
    for (int ct = 0; ct < 4; ++ct) {
        int c = ct * 16 + lcol;
        float* op = out + ((size_t)b * C_ + c) * N_ + n0 + wave * 16 + lhalf * 8;
        float4 v0 = { o[ct][0] * inv[0], o[ct][1] * inv[1], o[ct][2] * inv[2], o[ct][3] * inv[3] };
        float4 v1 = { o[ct][4] * inv[4], o[ct][5] * inv[5], o[ct][6] * inv[6], o[ct][7] * inv[7] };
        *(float4*)op       = v0;
        *(float4*)(op + 4) = v1;
    }
}

// ---------------------------------------------------------------------------
// Host launch. Workspace: qT (4MB) | kT (4MB) | vCN (4MB), all bf16.
// ---------------------------------------------------------------------------
extern "C" void kernel_launch(void* const* d_in, const int* in_sizes, int n_in,
                              void* d_out, int out_size, void* d_ws, size_t ws_size,
                              hipStream_t stream) {
    (void)in_sizes; (void)n_in; (void)out_size; (void)ws_size;
    const float* x  = (const float*)d_in[0];
    const float* Wq = (const float*)d_in[1];
    const float* bq = (const float*)d_in[2];
    const float* Wk = (const float*)d_in[3];
    const float* bk = (const float*)d_in[4];
    const float* Wv = (const float*)d_in[5];
    const float* bv = (const float*)d_in[6];

    __bf16* qT  = (__bf16*)d_ws;
    __bf16* kT  = qT + (size_t)B_ * N_ * C_;
    __bf16* vCN = kT + (size_t)B_ * N_ * C_;

    qkv_proj_kernel<<<B_ * (N_ / 64), 256, 0, stream>>>(x, Wq, bq, Wk, bk, Wv, bv,
                                                        qT, kT, vCN);
    flash_attn_kernel<<<B_ * (N_ / 64), 128, 0, stream>>>(qT, kT, vCN, (float*)d_out);
}
